// VQSignFeatures_55989193671248
// MI455X (gfx1250) — matrix operand order repaired and checked by reading, not compiled, not run on hardware
//
#include <hip/hip_runtime.h>
#include <hip/hip_bf16.h>
#include <math.h>

// ---------------------------------------------------------------------------
// Problem constants (from reference): B=32, T=512, D_IN=1024, D_CB=256,
// K_CB=8192, H_MID=512.
// ---------------------------------------------------------------------------
#define BB    32
#define TT    512
#define D_IN  1024
#define D_CB  256
#define K_CB  8192
#define H_MID 512
#define M_ALL (BB * TT)          // 16384 rows
#define M_GRU (BB * (TT - 1))    // 16352 rows (q_in / context)

typedef __attribute__((ext_vector_type(16))) __bf16 bf16x16;
typedef __attribute__((ext_vector_type(8)))  float  f32x8;

// ---------------------------------------------------------------------------
// fp32 -> bf16 cast (grid-stride)
// ---------------------------------------------------------------------------
__global__ void k_cast_bf16(const float* __restrict__ src, __bf16* __restrict__ dst, long n) {
    for (long i = (long)blockIdx.x * blockDim.x + threadIdx.x; i < n;
         i += (long)gridDim.x * blockDim.x)
        dst[i] = (__bf16)src[i];
}

// ---------------------------------------------------------------------------
// Row squared-norm of a [rows, 256] fp32 matrix
// ---------------------------------------------------------------------------
__global__ void k_rownorm256(const float* __restrict__ X, float* __restrict__ out, int rows) {
    int r = blockIdx.x * blockDim.x + threadIdx.x;
    if (r >= rows) return;
    const float* p = X + (long)r * 256;
    float s = 0.f;
    #pragma unroll 8
    for (int c = 0; c < 256; ++c) s += p[c] * p[c];
    out[r] = s;
}

__global__ void k_zero(float* p, int n) {
    int i = blockIdx.x * blockDim.x + threadIdx.x;
    if (i < n) p[i] = 0.f;
}

// ---------------------------------------------------------------------------
// Generic WMMA GEMM, register-blocked 16x64 per wave:
//   C[M,N] = act( A[M,K] @ W[N,K]^T + bias[N] )
// Per K-step: 1 A-fragment load, 4 B-fragment loads, 4 v_wmma (4x A reuse).
// remapT: if >0, logical row m maps to source row (m/(remapT-1))*remapT +
//         m%(remapT-1)  (used for q_in = quantized[:, :-1]).
// Requires N % 64 == 0, M % 16 == 0, K % 32 == 0.
// ---------------------------------------------------------------------------
__global__ void __launch_bounds__(256)
k_gemm(const __bf16* __restrict__ A, const __bf16* __restrict__ W,
       const float* __restrict__ bias,
       float* __restrict__ Cf, __bf16* __restrict__ Cb,
       int M, int N, int K, int relu, int remapT)
{
    const int wave = threadIdx.x >> 5;
    const int lane = threadIdx.x & 31;
    const int ngrp = N >> 6;                       // 64-column groups
    const long tile  = (long)blockIdx.x * (blockDim.x >> 5) + wave;
    const long total = (long)(M >> 4) * ngrp;
    if (tile >= total) return;
    const int row0 = (int)(tile / ngrp) << 4;
    const int col0 = (int)(tile % ngrp) << 6;

    // A fragment source row (ISA 7.12.2: 16-bit A 16x32 layout)
    int mA = row0 + (lane & 15);
    if (remapT > 0) mA = (mA / (remapT - 1)) * remapT + (mA % (remapT - 1));
    const __bf16* Arow = A + (long)mA * K + ((lane & 16) ? 8 : 0);
    // B fragments: column n of W^T is row n of W (contiguous K)
    const int nl = lane & 15;
    const __bf16* W0 = W + (long)(col0 + nl) * K + ((lane & 16) ? 8 : 0);
    const long wstep = (long)16 * K;               // 16 columns apart

    f32x8 acc[4] = {{}, {}, {}, {}};
    for (int k0 = 0; k0 < K; k0 += 32) {
        __builtin_prefetch(W0 + k0 + 128, 0, 1);   // global_prefetch_b8
        bf16x16 a;
        #pragma unroll
        for (int j = 0; j < 16; ++j) a[j] = Arow[k0 + ((j & 8) << 1) + (j & 7)];
        bf16x16 b0, b1, b2, b3;
        #pragma unroll
        for (int j = 0; j < 16; ++j) b0[j] = W0[k0 + ((j & 8) << 1) + (j & 7)];
        #pragma unroll
        for (int j = 0; j < 16; ++j) b1[j] = W0[wstep + k0 + ((j & 8) << 1) + (j & 7)];
        #pragma unroll
        for (int j = 0; j < 16; ++j) b2[j] = W0[2 * wstep + k0 + ((j & 8) << 1) + (j & 7)];
        #pragma unroll
        for (int j = 0; j < 16; ++j) b3[j] = W0[3 * wstep + k0 + ((j & 8) << 1) + (j & 7)];
        acc[0] = __builtin_amdgcn_wmma_f32_16x16x32_bf16(false, a, false, b0,
                                                         (short)0, acc[0], false, false);
        acc[1] = __builtin_amdgcn_wmma_f32_16x16x32_bf16(false, a, false, b1,
                                                         (short)0, acc[1], false, false);
        acc[2] = __builtin_amdgcn_wmma_f32_16x16x32_bf16(false, a, false, b2,
                                                         (short)0, acc[2], false, false);
        acc[3] = __builtin_amdgcn_wmma_f32_16x16x32_bf16(false, a, false, b3,
                                                         (short)0, acc[3], false, false);
    }

    const int mbase = row0 + ((lane & 16) ? 8 : 0);
    #pragma unroll
    for (int c = 0; c < 4; ++c) {
        const int n = col0 + c * 16 + nl;
        const float bv = bias ? bias[n] : 0.f;
        #pragma unroll
        for (int r = 0; r < 8; ++r) {
            float v = acc[c][r] + bv;
            if (relu) v = v > 0.f ? v : 0.f;
            const long off = (long)(mbase + r) * N + n;
            if (Cf) Cf[off] = v;
            if (Cb) Cb[off] = (__bf16)v;
        }
    }
}

// ---------------------------------------------------------------------------
// Conv1d(k=3, SAME, per-batch T=512) as 3 shifted WMMA GEMMs.
// A = h2 [M_ALL, 256] bf16, CW = conv_w [co][ci][3] bf16.
// ---------------------------------------------------------------------------
__global__ void __launch_bounds__(256)
k_conv(const __bf16* __restrict__ A, const __bf16* __restrict__ CW,
       const float* __restrict__ bias,
       float* __restrict__ Cf, __bf16* __restrict__ Cb)
{
    const int N = 256, K = 256;
    const int wave = threadIdx.x >> 5;
    const int lane = threadIdx.x & 31;
    const int ntiles = N >> 4;
    const long tile  = (long)blockIdx.x * 8 + wave;
    const long total = (long)(M_ALL >> 4) * ntiles;
    if (tile >= total) return;
    const int row0 = (int)(tile / ntiles) << 4;
    const int col0 = (int)(tile % ntiles) << 4;

    const int m  = row0 + (lane & 15);
    const int t  = m & (TT - 1);          // TT power of two
    const int khalf = (lane & 16) ? 8 : 0;
    const int n  = col0 + (lane & 15);
    const int kb = (lane & 16) ? 8 : 0;

    f32x8 acc = {};
    #pragma unroll
    for (int tap = 0; tap < 3; ++tap) {
        const int dt = tap - 1;
        const bool valid = ((unsigned)(t + dt) < (unsigned)TT);
        const __bf16* Arow = A + (long)(m + dt) * K + khalf;
        const __bf16* Wp   = CW + ((long)n * K + kb) * 3 + tap;   // stride 3 over ci
        for (int k0 = 0; k0 < K; k0 += 32) {
            bf16x16 a = {}, b;
            if (valid) {
                #pragma unroll
                for (int j = 0; j < 16; ++j) a[j] = Arow[k0 + ((j & 8) << 1) + (j & 7)];
            }
            #pragma unroll
            for (int j = 0; j < 16; ++j) b[j] = Wp[(long)(k0 + ((j & 8) << 1) + (j & 7)) * 3];
            acc = __builtin_amdgcn_wmma_f32_16x16x32_bf16(false, a, false, b,
                                                          (short)0, acc, false, false);
        }
    }
    const float bv = bias[n];
    const int mbase = row0 + ((lane & 16) ? 8 : 0);
    #pragma unroll
    for (int r = 0; r < 8; ++r) {
        float v = acc[r] + bv;
        const long off = (long)(mbase + r) * N + n;
        Cf[off] = v;
        Cb[off] = (__bf16)v;
    }
}

// ---------------------------------------------------------------------------
// VQ: fused distance-GEMM + argmin + gather + commitment-loss partial.
// Block = 256 threads (8 waves) handles 16 rows vs all 8192 codes.
// d2 = ||f||^2 - 2 f.c + ||c||^2 ; dot via WMMA, A-frags registered once.
// ---------------------------------------------------------------------------
__global__ void __launch_bounds__(256)
k_vq(const __bf16* __restrict__ Fbf, const __bf16* __restrict__ CBbf,
     const float* __restrict__ CB, const float* __restrict__ cbn,
     const float* __restrict__ rown, const float* __restrict__ Ff,
     float* __restrict__ idx_out, float* __restrict__ q_out,
     __bf16* __restrict__ q_bf, float* __restrict__ commit_accum)
{
    __shared__ __bf16 sA[16][256];
    __shared__ float  sMin[16][128];
    __shared__ int    sIdx[16][128];
    __shared__ int    sSel[16];

    const int tid  = threadIdx.x;
    const int wave = tid >> 5;
    const int lane = tid & 31;
    const int m0   = blockIdx.x * 16;

    for (int e = tid; e < 16 * 256; e += 256)
        sA[e >> 8][e & 255] = Fbf[(long)(m0 + (e >> 8)) * 256 + (e & 255)];
    __syncthreads();

    // Hoist the 8 A-fragments (K=256) into registers once per wave.
    bf16x16 afr[8];
    const int am    = lane & 15;
    const int khalf = (lane & 16) ? 8 : 0;
    #pragma unroll
    for (int kk = 0; kk < 8; ++kk)
        #pragma unroll
        for (int j = 0; j < 16; ++j)
            afr[kk][j] = sA[am][kk * 32 + khalf + ((j & 8) << 1) + (j & 7)];

    const int mb = (lane & 16) ? 8 : 0;
    float vmin[8]; int vidx[8]; float rn[8];
    #pragma unroll
    for (int r = 0; r < 8; ++r) { vmin[r] = 3.4e38f; vidx[r] = 0; rn[r] = rown[m0 + mb + r]; }

    // Each wave scans 512/8 = 64 codebook tiles.
    for (int nt = wave; nt < (K_CB >> 4); nt += 8) {
        const int n = (nt << 4) + (lane & 15);
        const __bf16* Bp = CBbf + (long)n * 256 + ((lane & 16) ? 8 : 0);
        __builtin_prefetch(Bp + 2048, 0, 1);
        f32x8 acc = {};
        #pragma unroll
        for (int kk = 0; kk < 8; ++kk) {
            bf16x16 b;
            #pragma unroll
            for (int j = 0; j < 16; ++j) b[j] = Bp[kk * 32 + ((j & 8) << 1) + (j & 7)];
            acc = __builtin_amdgcn_wmma_f32_16x16x32_bf16(false, afr[kk], false, b,
                                                          (short)0, acc, false, false);
        }
        const float cn = cbn[n];
        #pragma unroll
        for (int r = 0; r < 8; ++r) {
            const float d2 = rn[r] - 2.f * acc[r] + cn;
            if (d2 < vmin[r]) { vmin[r] = d2; vidx[r] = n; }
        }
    }

    const int slot = wave * 16 + (lane & 15);
    #pragma unroll
    for (int r = 0; r < 8; ++r) { sMin[mb + r][slot] = vmin[r]; sIdx[mb + r][slot] = vidx[r]; }
    __syncthreads();

    if (tid < 16) {
        float best = 3.4e38f; int bi = 0;
        for (int s = 0; s < 128; ++s) {
            const float v = sMin[tid][s];
            if (v < best) { best = v; bi = sIdx[tid][s]; }
        }
        sSel[tid] = bi;
        idx_out[m0 + tid] = (float)bi;
    }
    __syncthreads();

    float lsum = 0.f;
    for (int e = tid; e < 16 * 256; e += 256) {
        const int mm = e >> 8, c = e & 255;
        const float q = CB[(long)sSel[mm] * 256 + c];
        const long off = (long)(m0 + mm) * 256 + c;
        q_out[off] = q;
        q_bf[off]  = (__bf16)q;
        const float d = Ff[off] - q;
        lsum += d * d;
    }
    atomicAdd(commit_accum, lsum);
}

// ---------------------------------------------------------------------------
// GRU scan: single workgroup, 16 waves. h kept bf16 in LDS; gate pre-acts
// (32x768 fp32) in dynamic LDS (112 KB total -- legal on 320 KB WGP LDS).
// Per step: gh = h @ Whh^T + bhh. Wave w owns row-tile (w&1) and 6 column
// tiles [ (w>>1)*96 .. +96 ): A-frags hoisted from LDS once, B streamed.
// ---------------------------------------------------------------------------
__global__ void __launch_bounds__(512)
k_gru(const float* __restrict__ xproj, const __bf16* __restrict__ Whh,
      const float* __restrict__ bhh, float* __restrict__ ctx)
{
    extern __shared__ char smem[];
    __bf16* sh  = (__bf16*)smem;                       // 32*256 bf16 = 16 KB
    float*  sgh = (float*)(smem + 32 * 256 * 2);       // 32*768 f32  = 96 KB

    const int tid  = threadIdx.x;
    const int wave = tid >> 5;
    const int lane = tid & 31;

    const int row0  = (wave & 1) << 4;                 // 0 or 16
    const int cbase = (wave >> 1) * 96;                // 8 groups of 6 tiles
    const int am    = row0 + (lane & 15);
    const int khalf = (lane & 16) ? 8 : 0;
    const int mbase = row0 + ((lane & 16) ? 8 : 0);

    for (int e = tid; e < 32 * 256; e += 512) sh[e] = (__bf16)0.f;
    __syncthreads();

    for (int t = 0; t < TT - 1; ++t) {
        // Hoist this wave's 8 A-fragments (its 16 h-rows, K=256) once.
        bf16x16 afr[8];
        #pragma unroll
        for (int kk = 0; kk < 8; ++kk)
            #pragma unroll
            for (int j = 0; j < 16; ++j)
                afr[kk][j] = sh[am * 256 + kk * 32 + khalf + ((j & 8) << 1) + (j & 7)];

        // --- gh = h @ Whh^T + bhh : 6 column tiles, B streamed from L2 ---
        #pragma unroll 1
        for (int ct = 0; ct < 6; ++ct) {
            const int n = cbase + ct * 16 + (lane & 15);
            const __bf16* Bp = Whh + (long)n * 256 + ((lane & 16) ? 8 : 0);
            f32x8 acc = {};
            #pragma unroll
            for (int kk = 0; kk < 8; ++kk) {
                bf16x16 b;
                #pragma unroll
                for (int j = 0; j < 16; ++j) b[j] = Bp[kk * 32 + ((j & 8) << 1) + (j & 7)];
                acc = __builtin_amdgcn_wmma_f32_16x16x32_bf16(false, afr[kk], false, b,
                                                              (short)0, acc, false, false);
            }
            const float bv = bhh[n];
            #pragma unroll
            for (int r = 0; r < 8; ++r)
                sgh[(mbase + r) * 768 + n] = acc[r] + bv;
        }
        __syncthreads();

        // --- gates: each thread owns 16 of the 8192 h elements ---
        float hnew[16];
        #pragma unroll
        for (int q = 0; q < 16; ++q) {
            const int e = tid + q * 512;
            const int b = e >> 8, i = e & 255;
            const long xo = ((long)(b * (TT - 1) + t)) * 768 + i;
            const float r  = 1.f / (1.f + expf(-(xproj[xo]       + sgh[b * 768 + i])));
            const float z  = 1.f / (1.f + expf(-(xproj[xo + 256] + sgh[b * 768 + i + 256])));
            const float nn = tanhf(xproj[xo + 512] + r * sgh[b * 768 + i + 512]);
            const float hp = (float)sh[b * 256 + i];
            hnew[q] = (1.f - z) * nn + z * hp;
            ctx[((long)(b * (TT - 1) + t)) * 256 + i] = hnew[q];
        }
        __syncthreads();
        #pragma unroll
        for (int q = 0; q < 16; ++q) sh[tid + q * 512] = (__bf16)hnew[q];
        __syncthreads();
    }
}

// ---------------------------------------------------------------------------
// context loss partial:  sum (ctx[b,t,:] - features[b,t+1,:])^2
// ---------------------------------------------------------------------------
__global__ void k_ctxloss(const float* __restrict__ ctx, const float* __restrict__ Ff,
                          float* __restrict__ accum)
{
    const long n = (long)M_GRU * 256;
    float ls = 0.f;
    for (long e = (long)blockIdx.x * blockDim.x + threadIdx.x; e < n;
         e += (long)gridDim.x * blockDim.x) {
        const long m = e >> 8; const int c = (int)(e & 255);
        const long b = m / (TT - 1), t = m % (TT - 1);
        const float d = ctx[e] - Ff[((b * TT) + t + 1) * 256 + c];
        ls += d * d;
    }
    atomicAdd(accum, ls);
}

__global__ void k_finalize(const float* __restrict__ accum, float* __restrict__ out) {
    if (threadIdx.x == 0 && blockIdx.x == 0) {
        const float cl = accum[0] / ((float)M_ALL * 256.f);   // commitment == codebook numerically
        const float xl = accum[1] / ((float)M_GRU * 256.f);
        out[0] = cl;
        out[1] = cl;
        out[2] = xl;
        out[3] = cl + 0.25f * cl + 0.1f * xl;
    }
}

// ---------------------------------------------------------------------------
// Host orchestration
// ---------------------------------------------------------------------------
static inline int gemm_blocks64(int M, int N) { return ((M >> 4) * (N >> 6) + 7) / 8; }
static inline int gemm_blocks16(int M, int N) { return ((M >> 4) * (N >> 4) + 7) / 8; }

extern "C" void kernel_launch(void* const* d_in, const int* in_sizes, int n_in,
                              void* d_out, int out_size, void* d_ws, size_t ws_size,
                              hipStream_t stream) {
    (void)in_sizes; (void)n_in; (void)out_size; (void)ws_size;

    const float* x      = (const float*)d_in[0];
    const float* w1     = (const float*)d_in[1];
    const float* b1     = (const float*)d_in[2];
    const float* w2     = (const float*)d_in[3];
    const float* b2     = (const float*)d_in[4];
    const float* conv_w = (const float*)d_in[5];
    const float* conv_b = (const float*)d_in[6];
    const float* cb     = (const float*)d_in[7];
    const float* wih    = (const float*)d_in[8];
    const float* whh    = (const float*)d_in[9];
    const float* bih    = (const float*)d_in[10];
    const float* bhh    = (const float*)d_in[11];

    float* out      = (float*)d_out;
    float* idx_out  = out;                              // 16384
    float* q_out    = out + M_ALL;                      // 16384*256
    float* loss_out = out + M_ALL + (long)M_ALL * D_CB; // 4 scalars

    char* ws = (char*)d_ws;
    auto carve = [&](size_t bytes) {
        char* p = ws;
        ws += (bytes + 255) & ~(size_t)255;
        return p;
    };
    __bf16* x_bf   = (__bf16*)carve((size_t)M_ALL * D_IN * 2);
    __bf16* w1_bf  = (__bf16*)carve((size_t)H_MID * D_IN * 2);
    __bf16* h1_bf  = (__bf16*)carve((size_t)M_ALL * H_MID * 2);
    __bf16* w2_bf  = (__bf16*)carve((size_t)D_CB * H_MID * 2);
    __bf16* h2_bf  = (__bf16*)carve((size_t)M_ALL * D_CB * 2);
    __bf16* cw_bf  = (__bf16*)carve((size_t)D_CB * D_CB * 3 * 2);
    float*  feat_f = (float*) carve((size_t)M_ALL * D_CB * 4);
    __bf16* feat_b = (__bf16*)carve((size_t)M_ALL * D_CB * 2);
    __bf16* cb_bf  = (__bf16*)carve((size_t)K_CB * D_CB * 2);
    float*  cb_n   = (float*) carve((size_t)K_CB * 4);
    float*  row_n  = (float*) carve((size_t)M_ALL * 4);
    __bf16* q_bf   = (__bf16*)carve((size_t)M_ALL * D_CB * 2);
    __bf16* wih_bf = (__bf16*)carve((size_t)3 * D_CB * D_CB * 2);
    __bf16* whh_bf = (__bf16*)carve((size_t)3 * D_CB * D_CB * 2);
    float*  xproj  = (float*) carve((size_t)M_GRU * 3 * D_CB * 4);
    float*  ctx    = (float*) carve((size_t)M_GRU * D_CB * 4);
    float*  accum  = (float*) carve(8 * 4);

    // --- precision staging (fp32 -> bf16) ---
    auto cast = [&](const float* s, __bf16* d, long n) {
        int blk = (int)((n + 255) / 256); if (blk > 16384) blk = 16384;
        k_cast_bf16<<<blk, 256, 0, stream>>>(s, d, n);
    };
    cast(x,      x_bf,   (long)M_ALL * D_IN);
    cast(w1,     w1_bf,  (long)H_MID * D_IN);
    cast(w2,     w2_bf,  (long)D_CB * H_MID);
    cast(conv_w, cw_bf,  (long)D_CB * D_CB * 3);
    cast(cb,     cb_bf,  (long)K_CB * D_CB);
    cast(wih,    wih_bf, (long)3 * D_CB * D_CB);
    cast(whh,    whh_bf, (long)3 * D_CB * D_CB);

    k_zero<<<1, 8, 0, stream>>>(accum, 8);

    // --- MLP1: h1 = relu(x @ w1^T + b1)  [16384,1024]x[1024,512] ---
    k_gemm<<<gemm_blocks64(M_ALL, H_MID), 256, 0, stream>>>(
        x_bf, w1_bf, b1, nullptr, h1_bf, M_ALL, H_MID, D_IN, 1, 0);

    // --- MLP2: h2 = relu(h1 @ w2^T + b2)  [16384,512]x[512,256] ---
    k_gemm<<<gemm_blocks64(M_ALL, D_CB), 256, 0, stream>>>(
        h1_bf, w2_bf, b2, nullptr, h2_bf, M_ALL, D_CB, H_MID, 1, 0);

    // --- conv1d(k=3, SAME) as 3 shifted WMMA GEMMs -> features ---
    k_conv<<<gemm_blocks16(M_ALL, D_CB), 256, 0, stream>>>(
        h2_bf, cw_bf, conv_b, feat_f, feat_b);

    // --- norms for the distance decomposition ---
    k_rownorm256<<<(M_ALL + 255) / 256, 256, 0, stream>>>(feat_f, row_n, M_ALL);
    k_rownorm256<<<(K_CB + 255) / 256, 256, 0, stream>>>(cb, cb_n, K_CB);

    // --- VQ: fused 16384x8192x256 distance GEMM + argmin + gather ---
    k_vq<<<M_ALL / 16, 256, 0, stream>>>(
        feat_b, cb_bf, cb, cb_n, row_n, feat_f, idx_out, q_out, q_bf, accum);

    // --- xproj = q_in @ wih^T + bih  (rows remapped to skip t=T-1) ---
    k_gemm<<<gemm_blocks64(M_GRU, 3 * D_CB), 256, 0, stream>>>(
        q_bf, wih_bf, bih, xproj, nullptr, M_GRU, 3 * D_CB, D_CB, 0, TT);

    // --- GRU scan (single WG, 112 KB dynamic LDS) ---
    const size_t gru_lds = (size_t)32 * 256 * 2 + (size_t)32 * 768 * 4;
    k_gru<<<1, 512, gru_lds, stream>>>(xproj, whh_bf, bhh, ctx);

    // --- losses ---
    k_ctxloss<<<2048, 256, 0, stream>>>(ctx, feat_f, accum + 1);
    k_finalize<<<1, 32, 0, stream>>>(accum, loss_out);
}
